// CausalSelfAttention_65171833749838
// MI455X (gfx1250) — matrix operand order repaired and checked by reading, not compiled, not run on hardware
//
#include <hip/hip_runtime.h>
#include <hip/hip_bf16.h>

typedef __attribute__((ext_vector_type(16))) _Float16 v16h;
typedef __attribute__((ext_vector_type(8)))  _Float16 v8h;
typedef __attribute__((ext_vector_type(8)))  float    v8f;
typedef __attribute__((ext_vector_type(4)))  int      v4i;

#define B_    2
#define T_    2048
#define C_    1024
#define H_    16
#define HS_   64
#define HALF_ 32
#define N3C_  3072
#define BT_   (B_*T_)

union VA { v16h v; v8h h[2]; };

// ---------------------------------------------------------------- async LDS copy
#if defined(__AMDGCN__)
# if __has_builtin(__builtin_amdgcn_global_load_async_to_lds_b128)
#  define ASYNC_COPY 1
# else
#  define ASYNC_COPY 0
#  warning "gfx1250 async-to-LDS builtin not available on device pass; using sync staging"
# endif
#else
# define ASYNC_COPY 0
#endif

__device__ __forceinline__ void cp_b128(void* lds, const void* g) {
#if ASYNC_COPY
  typedef __attribute__((address_space(1))) v4i GV;   // global int4
  typedef __attribute__((address_space(3))) v4i LV;   // LDS int4
  __builtin_amdgcn_global_load_async_to_lds_b128((GV*)g, (LV*)lds, 0, 0);
#else
  *(v8h*)lds = *(const v8h*)g;
#endif
}

__device__ __forceinline__ void cp_wait() {
#if ASYNC_COPY
# if __has_builtin(__builtin_amdgcn_s_wait_asynccnt)
  __builtin_amdgcn_s_wait_asynccnt(0);
# else
  asm volatile("s_wait_asynccnt 0" ::: "memory");
# endif
#endif
}

// ---------------------------------------------------------------- conversions
__global__ void k_cvt_f16(const float* __restrict__ src, _Float16* __restrict__ dst, int n) {
  int i = blockIdx.x * blockDim.x + threadIdx.x;
  if (i < n) dst[i] = (_Float16)src[i];
}

// src: K x N (row-major f32)  ->  dst: N x K (row-major f16)
__global__ void k_cvt_T(const float* __restrict__ src, _Float16* __restrict__ dst, int N, int K) {
  int i = blockIdx.x * blockDim.x + threadIdx.x;
  if (i < N * K) {
    int n = i / K, k = i - n * K;
    dst[i] = (_Float16)src[(long)k * N + n];
  }
}

// ---------------------------------------------------------------- WMMA GEMM
// Out[M][N] = A[M][K] * Wt[N][K]^T + bias[N]
// block tile: 128(M) x 128(N), BK = 32, 256 threads = 8 waves.
// wave w owns rows [w*16, w*16+16) x all 128 cols: 1 A-frag reused over 8 B-frags.
template <typename OutT>
__global__ __launch_bounds__(256) void k_gemm_wmma(const _Float16* __restrict__ A,
                                                   const _Float16* __restrict__ Wt,
                                                   const float* __restrict__ bias,
                                                   OutT* __restrict__ Out,
                                                   int M, int N, int K) {
  __shared__ _Float16 As[128 * 40];   // rows padded 32 -> 40 halves
  __shared__ _Float16 Bs[128 * 40];
  const int tid = threadIdx.x;
  const int w  = tid >> 5;
  const int L  = tid & 31;
  const int hl = L >> 4;        // half-of-wave (0/1)
  const int ln = L & 15;        // lane-in-half
  const int m0 = blockIdx.x * 128;
  const int n0 = blockIdx.y * 128;

  v8f z = {};
  v8f acc[8];
#pragma unroll
  for (int j = 0; j < 8; ++j) acc[j] = z;

  const int srow = tid >> 1, sc = (tid & 1) * 16;   // staging: 16 halves / thread / tile

  for (int k0 = 0; k0 < K; k0 += 32) {
    // stage A tile 128x32 and B tile 128x32 into LDS (async memory->LDS)
    {
      const _Float16* ga = A  + (long)(m0 + srow) * K + k0 + sc;
      cp_b128(&As[srow * 40 + sc],     ga);
      cp_b128(&As[srow * 40 + sc + 8], ga + 8);
      const _Float16* gb = Wt + (long)(n0 + srow) * K + k0 + sc;
      cp_b128(&Bs[srow * 40 + sc],     gb);
      cp_b128(&Bs[srow * 40 + sc + 8], gb + 8);
    }
    cp_wait();
    __syncthreads();

    // A fragment: 16x32 f16, ISA layout -> two ds_load_b128 per lane
    VA af;
    const _Float16* ab = &As[(w * 16 + ln) * 40 + hl * 8];
    af.h[0] = *(const v8h*)ab;          // k = hl*8 .. +7
    af.h[1] = *(const v8h*)(ab + 16);   // k = 16+hl*8 .. +7

#pragma unroll
    for (int j = 0; j < 8; ++j) {
      // B fragment: lane holds column n, K values hl*8..+7 and 16+hl*8..+7 (matches A)
      const _Float16* bb = &Bs[(j * 16 + ln) * 40 + hl * 8];
      VA bf; bf.h[0] = *(const v8h*)bb; bf.h[1] = *(const v8h*)(bb + 16);
      acc[j] = __builtin_amdgcn_wmma_f32_16x16x32_f16(false, af.v, false, bf.v,
                                                      (short)0, acc[j], false, false);
    }
    __syncthreads();
  }

#pragma unroll
  for (int j = 0; j < 8; ++j) {
    int col = n0 + j * 16 + ln;
    float bcol = bias[col];
#pragma unroll
    for (int r = 0; r < 8; ++r) {
      int row = m0 + w * 16 + r + hl * 8;
      Out[(long)row * N + col] = (OutT)(acc[j][r] + bcol);
    }
  }
}

// ---------------------------------------------------------------- RoPE + head packing
// qkv: [B*T][3072] f16 -> Qh,Kh: [B,H,T,HS] f16 (RoPE applied), Vt: [B,H,HS,T] f16
__global__ void k_rope_pack(const _Float16* __restrict__ qkv,
                            const float* __restrict__ cosp, const float* __restrict__ sinp,
                            _Float16* __restrict__ Qh, _Float16* __restrict__ Kh,
                            _Float16* __restrict__ Vt) {
  int i = blockIdx.x * blockDim.x + threadIdx.x;
  if (i >= B_ * T_ * H_ * HS_) return;
  int d = i & 63;
  int h = (i >> 6) & 15;
  int t = (i >> 10) & 2047;
  int b = i >> 21;
  int dd = d & 31;
  float c = cosp[t * HALF_ + dd], s = sinp[t * HALF_ + dd];
  long base = (long)(b * T_ + t) * N3C_ + h * HS_;
  float q1 = (float)qkv[base + dd],        q2 = (float)qkv[base + dd + 32];
  float k1 = (float)qkv[base + C_ + dd],   k2 = (float)qkv[base + C_ + dd + 32];
  float v  = (float)qkv[base + 2 * C_ + d];
  float qr = (d < 32) ? (q1 * c - q2 * s) : (q1 * s + q2 * c);
  float kr = (d < 32) ? (k1 * c - k2 * s) : (k1 * s + k2 * c);
  long hb = ((long)(b * H_ + h) * T_ + t) * HS_ + d;
  Qh[hb] = (_Float16)qr;
  Kh[hb] = (_Float16)kr;
  Vt[((long)(b * H_ + h) * HS_ + d) * T_ + t] = (_Float16)v;
}

// ---------------------------------------------------------------- flash attention
// grid: B*H*(T/128). 8 waves; wave w owns q rows [qb*128+w*16, +16). K/V tiles of 32 in LDS.
__global__ __launch_bounds__(256) void k_flash(const _Float16* __restrict__ Qh,
                                               const _Float16* __restrict__ Kh,
                                               const _Float16* __restrict__ Vt,
                                               _Float16* __restrict__ Ctx) {
  __shared__ _Float16 Ks[32 * 72];        // 32 k-rows x 64 d (padded to 72)
  __shared__ _Float16 Vs[64 * 40];        // 64 d-rows x 32 t (padded to 40)
  __shared__ _Float16 Ps[8 * 16 * 40];    // per-wave 16x32 P staging (padded)
  const int tid = threadIdx.x;
  const int w  = tid >> 5;
  const int L  = tid & 31;
  const int hl = L >> 4;
  const int ln = L & 15;
  const int qb = blockIdx.x & 15;
  const int h  = (blockIdx.x >> 4) & 15;
  const int b  = blockIdx.x >> 8;
  const int qw0 = qb * 128 + w * 16;
  const long bhT = (long)(b * H_ + h) * T_;
  const long bhV = (long)(b * H_ + h) * HS_;

  // Q fragments for k-dim halves [0,32) and [32,64): loaded once from global
  VA qf0, qf1;
  {
    const _Float16* qrow = Qh + (bhT + qw0 + ln) * HS_;
    qf0.h[0] = *(const v8h*)(qrow + hl * 8);
    qf0.h[1] = *(const v8h*)(qrow + 16 + hl * 8);
    qf1.h[0] = *(const v8h*)(qrow + 32 + hl * 8);
    qf1.h[1] = *(const v8h*)(qrow + 48 + hl * 8);
  }

  v8f z = {};
  v8f o[4]; o[0] = z; o[1] = z; o[2] = z; o[3] = z;
  float mi[8], li[8];
#pragma unroll
  for (int r = 0; r < 8; ++r) { mi[r] = -1e30f; li[r] = 0.0f; }

  const int krow = tid >> 3, kc = (tid & 7) * 8;
  const int vrow = tid >> 2, vc = (tid & 3) * 8;
  const int ktEnd = qb * 4 + 4;           // causal bound for this 128-row block
  const float scale = 0.125f;             // 1/sqrt(64)

  for (int kt = 0; kt < ktEnd; ++kt) {
    const int k0 = kt * 32;
    cp_b128(&Ks[krow * 72 + kc], Kh + (bhT + k0 + krow) * HS_ + kc);
    cp_b128(&Vs[vrow * 40 + vc], Vt + (bhV + vrow) * T_ + k0 + vc);
    if (kt + 1 < ktEnd) {                 // prefetch next tile into L2/WGP$
      __builtin_prefetch(Kh + (bhT + k0 + 32 + krow) * HS_ + kc, 0, 0);
      __builtin_prefetch(Vt + (bhV + vrow) * T_ + k0 + 32 + vc, 0, 0);
    }
    cp_wait();
    __syncthreads();

    if (k0 <= qw0 + 15) {                 // wave-level causal skip
      float p[2][8];
#pragma unroll
      for (int j = 0; j < 2; ++j) {
        v8f sa = z;
        const _Float16* kb = &Ks[(j * 16 + ln) * 72];
        VA b0, b1;
        b0.h[0] = *(const v8h*)(kb + hl * 8);      b0.h[1] = *(const v8h*)(kb + 16 + hl * 8);
        b1.h[0] = *(const v8h*)(kb + 32 + hl * 8); b1.h[1] = *(const v8h*)(kb + 48 + hl * 8);
        sa = __builtin_amdgcn_wmma_f32_16x16x32_f16(false, qf0.v, false, b0.v, (short)0, sa, false, false);
        sa = __builtin_amdgcn_wmma_f32_16x16x32_f16(false, qf1.v, false, b1.v, (short)0, sa, false, false);
#pragma unroll
        for (int r = 0; r < 8; ++r) p[j][r] = sa[r] * scale;
      }
      if (k0 + 31 > qw0) {                // diagonal tile: apply causal mask
#pragma unroll
        for (int j = 0; j < 2; ++j) {
          int kg = k0 + j * 16 + ln;
#pragma unroll
          for (int r = 0; r < 8; ++r) {
            int qg = qw0 + r + hl * 8;
            if (kg > qg) p[j][r] = -1e30f;
          }
        }
      }
      // online softmax; row m = r + 8*hl lives across the 16 lanes of half hl
#pragma unroll
      for (int r = 0; r < 8; ++r) {
        float rm = fmaxf(p[0][r], p[1][r]);
#pragma unroll
        for (int off = 1; off <= 8; off <<= 1)
          rm = fmaxf(rm, __shfl_xor(rm, off, 32));
        float mnew = fmaxf(mi[r], rm);
        float alpha = __expf(mi[r] - mnew);
        mi[r] = mnew;
        p[0][r] = __expf(p[0][r] - mnew);
        p[1][r] = __expf(p[1][r] - mnew);
        float rs = p[0][r] + p[1][r];
#pragma unroll
        for (int off = 1; off <= 8; off <<= 1)
          rs += __shfl_xor(rs, off, 32);
        li[r] = alpha * li[r] + rs;
        o[0][r] *= alpha; o[1][r] *= alpha; o[2][r] *= alpha; o[3][r] *= alpha;
      }
      // C-layout -> A-layout for P via per-wave LDS round trip
      _Float16* pw = &Ps[w * 16 * 40];
#pragma unroll
      for (int j = 0; j < 2; ++j)
#pragma unroll
        for (int r = 0; r < 8; ++r)
          pw[(r + hl * 8) * 40 + j * 16 + ln] = (_Float16)p[j][r];
      asm volatile("s_wait_dscnt 0" ::: "memory");   // CDNA5 split DS counter wait
      VA pf;
      const _Float16* pr = pw + ln * 40 + hl * 8;
      pf.h[0] = *(const v8h*)pr;
      pf.h[1] = *(const v8h*)(pr + 16);
#pragma unroll
      for (int j2 = 0; j2 < 4; ++j2) {
        const _Float16* vb = &Vs[(j2 * 16 + ln) * 40 + hl * 8];
        VA vf; vf.h[0] = *(const v8h*)vb; vf.h[1] = *(const v8h*)(vb + 16);
        o[j2] = __builtin_amdgcn_wmma_f32_16x16x32_f16(false, pf.v, false, vf.v, (short)0, o[j2], false, false);
      }
    }
    __syncthreads();
  }

  // normalize and write ctx back as [B,T,C] f16
#pragma unroll
  for (int j2 = 0; j2 < 4; ++j2) {
    int d = j2 * 16 + ln;
#pragma unroll
    for (int r = 0; r < 8; ++r) {
      int t = qw0 + r + hl * 8;
      float val = o[j2][r] / li[r];
      Ctx[(long)(b * T_ + t) * C_ + h * HS_ + d] = (_Float16)val;
    }
  }
}

// ---------------------------------------------------------------- launch
extern "C" void kernel_launch(void* const* d_in, const int* in_sizes, int n_in,
                              void* d_out, int out_size, void* d_ws, size_t ws_size,
                              hipStream_t stream) {
  const float* x      = (const float*)d_in[0];
  const float* cosp   = (const float*)d_in[1];
  const float* sinp   = (const float*)d_in[2];
  const float* w_attn = (const float*)d_in[3];
  const float* b_attn = (const float*)d_in[4];
  const float* w_proj = (const float*)d_in[5];
  const float* b_proj = (const float*)d_in[6];
  float* out = (float*)d_out;

  char* ws = (char*)d_ws;
  _Float16* Xh  = (_Float16*)(ws);                 //  8 MB: x in f16
  _Float16* WaT = (_Float16*)(ws + 8388608);       //  6 MB: w_attn^T f16 (3072 x 1024)
  _Float16* WpT = (_Float16*)(ws + 14680064);      //  2 MB: w_proj^T f16 (1024 x 1024)
  _Float16* QKV = (_Float16*)(ws + 16777216);      // 24 MB: qkv f16 (4096 x 3072)
  _Float16* Qh  = (_Float16*)(ws + 41943040);      //  8 MB
  _Float16* Kh  = (_Float16*)(ws + 50331648);      //  8 MB
  _Float16* Vt  = (_Float16*)(ws + 58720256);      //  8 MB
  _Float16* Ctx = (_Float16*)(ws + 67108864);      //  8 MB

  k_cvt_f16<<<(BT_ * C_) / 256, 256, 0, stream>>>(x, Xh, BT_ * C_);
  k_cvt_T<<<(N3C_ * C_) / 256, 256, 0, stream>>>(w_attn, WaT, N3C_, C_);
  k_cvt_T<<<(C_ * C_) / 256, 256, 0, stream>>>(w_proj, WpT, C_, C_);

  dim3 g1(BT_ / 128, N3C_ / 128);
  k_gemm_wmma<_Float16><<<g1, 256, 0, stream>>>(Xh, WaT, b_attn, QKV, BT_, N3C_, C_);

  k_rope_pack<<<(B_ * T_ * H_ * HS_) / 256, 256, 0, stream>>>(QKV, cosp, sinp, Qh, Kh, Vt);

  k_flash<<<B_ * H_ * (T_ / 128), 256, 0, stream>>>(Qh, Kh, Vt, Ctx);

  dim3 g2(BT_ / 128, C_ / 128);
  k_gemm_wmma<float><<<g2, 256, 0, stream>>>(Ctx, WpT, b_proj, out, BT_, C_, C_);
}
